// RetBlock_11184094838862
// MI455X (gfx1250) — matrix operand order repaired and checked by reading, not compiled, not run on hardware
//
#include <hip/hip_runtime.h>
#include <hip/hip_bf16.h>
#include <math.h>

// ---------------------------------------------------------------------------
// CDNA5 (gfx1250) wave32 WMMA implementation of the 2D retention block.
// Matmuls: v_wmma_f32_16x16x32_bf16. GEMM tile fills: global_load_async_to_lds.
// Weights are pre-transposed to bf16 Wt[N][K] so WMMA B-frags are contiguous.
// ---------------------------------------------------------------------------

typedef __attribute__((ext_vector_type(16))) __bf16 v16bf;
typedef __attribute__((ext_vector_type(8)))  __bf16 v8bf;
typedef __attribute__((ext_vector_type(8)))  float  v8f;

// round-to-nearest-even f32 -> bf16
static __device__ inline __bf16 f2bf(float f) {
    unsigned u = __builtin_bit_cast(unsigned, f);
    unsigned r = (u + 0x7FFFu + ((u >> 16) & 1u)) >> 16;
    unsigned short s = (unsigned short)r;
    return __builtin_bit_cast(__bf16, s);
}

static __device__ inline v16bf cat8(v8bf lo, v8bf hi) {
    return __builtin_shufflevector(lo, hi, 0,1,2,3,4,5,6,7,8,9,10,11,12,13,14,15);
}

// Async 16B global -> LDS copy (CDNA5 Tensor/Async path, ASYNCcnt tracked).
// GV addressing: 64-bit global address in VGPR pair, LDS byte addr in VGPR.
static __device__ inline void async_copy_b128(const void* gptr, void* lptr) {
    unsigned lds = (unsigned)(size_t)lptr;     // LDS aperture: addr[31:0] = offset
    asm volatile("global_load_async_to_lds_b128 %0, %1, off"
                 :: "v"(lds), "v"(gptr) : "memory");
}
static __device__ inline void async_wait0() {
    asm volatile("s_wait_asynccnt 0" ::: "memory");
}

// A fragment (16x32 bf16, row-major LDS tile, stride in elements).
static __device__ inline v16bf lds_frag_a(const __bf16* base, int row0, int kbase,
                                          int stride, int lane) {
    int r  = row0 + (lane & 15);
    int kh = (lane >> 4) << 3;                 // 0 or 8
    const __bf16* p = base + r * stride + kbase + kh;
    v8bf lo = *(const v8bf*)(p);
    v8bf hi = *(const v8bf*)(p + 16);
    return cat8(lo, hi);
}

// B fragment (32x16 bf16) from a COLUMN-MAJOR LDS tile Bt[n][k].
static __device__ inline v16bf lds_frag_b(const __bf16* base, int ncol, int kbase,
                                          int stride, int lane) {
    int n  = ncol + (lane & 15);
    int ks = kbase + ((lane >> 4) << 3);       // 0 or 8
    const __bf16* p = base + n * stride + ks;
    v8bf lo = *(const v8bf*)(p);
    v8bf hi = *(const v8bf*)(p + 16);
    return cat8(lo, hi);
}

static __device__ inline float gelu_exact(float x) {
    return 0.5f * x * (1.0f + erff(x * 0.70710678118654752f));
}

// ---------------------------------------------------------------------------
// bf16 WMMA GEMM: C[M,N] = A[M,K] * Wt[N,K]^T (+ epilogue)
// A row-major [M,K]; weights pre-transposed col-major Wt[N][K].
// Block: 128 threads = 4 waves; block tile 64x64; wave tile 32x32 (2x2 WMMA).
// EPI 0: outB = bf16(acc + bias)                          (QKV)
// EPI 1: outF = resid + (acc + bias) * gamma              (proj + residual)
// EPI 2: outB = bf16(gelu(acc + bias))                    (FFN1)
// EPI 3: outF = resid + acc + bias                        (FFN2 + residual)
// ---------------------------------------------------------------------------
template <int EPI>
__global__ __launch_bounds__(128) void gemm_bf16_wmma(
    const __bf16* __restrict__ A, const __bf16* __restrict__ Wt,
    const float* __restrict__ bias, const float* __restrict__ resid,
    const float* __restrict__ gamma,
    float* __restrict__ outF, __bf16* __restrict__ outB,
    int M, int N, int K)
{
    constexpr int LDT = 40;                    // padded LDS stride (bf16 elems)
    __shared__ __align__(16) __bf16 As[64 * LDT];
    __shared__ __align__(16) __bf16 Bt[64 * LDT];   // col-major tile: Bt[n][k]

    const int tid  = threadIdx.x;
    const int lane = tid & 31;
    const int wv   = tid >> 5;                 // 0..3
    const int wm   = (wv & 1) * 32;
    const int wn   = (wv >> 1) * 32;
    const int bm   = blockIdx.x * 64;
    const int bn   = blockIdx.y * 64;

    v8f acc[2][2] = {};

    for (int k0 = 0; k0 < K; k0 += 32) {
        // --- A tile 64x32 and B tile 64x32 (both 16B-contiguous): async -> LDS
        #pragma unroll
        for (int p = 0; p < 2; ++p) {
            int e = (tid + p * 128) * 8;       // 0..2040
            int r = e >> 5, c = e & 31;
            async_copy_b128(A  + (size_t)(bm + r) * K + k0 + c, As + r * LDT + c);
            async_copy_b128(Wt + (size_t)(bn + r) * K + k0 + c, Bt + r * LDT + c);
        }
        async_wait0();
        __syncthreads();

        #pragma unroll
        for (int mi = 0; mi < 2; ++mi) {
            v16bf a = lds_frag_a(As, wm + mi * 16, 0, LDT, lane);
            #pragma unroll
            for (int ni = 0; ni < 2; ++ni) {
                v16bf b = lds_frag_b(Bt, wn + ni * 16, 0, LDT, lane);
                acc[mi][ni] = __builtin_amdgcn_wmma_f32_16x16x32_bf16(
                    false, a, false, b, (short)0, acc[mi][ni], false, false);
            }
        }
        __syncthreads();                       // all reads done before next async fill
    }

    // --- epilogue: C/D layout -> lane (n = lane&15), VGPR i -> m = i + 8*(lane>=16)
    const int ln = lane & 15;
    const int lm = (lane >> 4) * 8;
    #pragma unroll
    for (int mi = 0; mi < 2; ++mi)
    #pragma unroll
    for (int ni = 0; ni < 2; ++ni)
    #pragma unroll
    for (int i = 0; i < 8; ++i) {
        int gm = bm + wm + mi * 16 + lm + i;
        int gn = bn + wn + ni * 16 + ln;
        size_t idx = (size_t)gm * N + gn;
        float v = acc[mi][ni][i] + bias[gn];
        if constexpr (EPI == 0) {
            outB[idx] = f2bf(v);
        } else if constexpr (EPI == 1) {
            outF[idx] = resid[idx] + v * gamma[gn];
        } else if constexpr (EPI == 2) {
            outB[idx] = f2bf(gelu_exact(v));
        } else {
            outF[idx] = resid[idx] + v;
        }
    }
}

// ---------------------------------------------------------------------------
// Retention, one axis. One block per (b, head, fixed-coordinate).
// mode 0: row (height) attention, fixed w, sequence over h (decay_h)
// mode 1: col (width)  attention, fixed h, sequence over w (decay_w)
// head_dim = 32 == WMMA K  ->  QK^T score tile is a single WMMA.
// Sequence 80 = 5 x 16 tiles; S.V contraction padded 80 -> 96 (3 K-steps).
// Block: 160 threads = 5 waves; wave i owns score/output rows [16i,16i+16).
// ---------------------------------------------------------------------------
__global__ __launch_bounds__(160) void retention_axis(
    const __bf16* __restrict__ qkv, const float* __restrict__ decay,
    float* __restrict__ out, int mode)
{
    constexpr int LQK = 40;    // stride for Q/K tiles (K=32 padded)
    constexpr int LSV = 104;   // stride for S (96-pad) and V^T tiles
    __shared__ __align__(16) __bf16 Qs[80 * LQK];
    __shared__ __align__(16) __bf16 Ks[80 * LQK];   // row-major == col-major B of Q.K^T
    __shared__ __align__(16) __bf16 Vt[32 * LSV];   // V transposed: Vt[c][pos], pos padded to 96
    __shared__ __align__(16) __bf16 Ss[80 * LSV];   // decayed scores, cols padded to 96

    const int tid  = threadIdx.x;
    const int lane = tid & 31;
    const int wv   = tid >> 5;                 // 0..4
    int bid = blockIdx.x;
    const int fix  = bid % 80;  bid /= 80;
    const int head = bid % 8;
    const int b    = bid / 8;

    const __bf16 bz = f2bf(0.0f);

    // load Q,K,V head slices [80 x 32] (already bf16 from the QKV GEMM)
    for (int idx = tid; idx < 80 * 32; idx += 160) {
        int s = idx >> 5, c = idx & 31;
        int row = (mode == 0) ? ((b * 80 + s) * 80 + fix)
                              : ((b * 80 + fix) * 80 + s);
        const __bf16* p = qkv + (size_t)row * 768 + head * 32 + c;
        Qs[s * LQK + c] = p[0];
        Ks[s * LQK + c] = p[256];
        Vt[c * LSV + s] = p[512];
    }
    for (int idx = tid; idx < 32 * 16; idx += 160)             // zero V pad rows 80..95
        Vt[(idx >> 4) * LSV + 80 + (idx & 15)] = bz;
    __syncthreads();

    const float* dh = decay + (size_t)head * 80 * 80;

    // S = (Q K^T) * decay for this wave's 16 query rows; K-dim = 32 -> 1 WMMA/tile
    {
        v16bf a = lds_frag_a(Qs, wv * 16, 0, LQK, lane);
        #pragma unroll
        for (int nt = 0; nt < 5; ++nt) {
            v16bf kb = lds_frag_b(Ks, nt * 16, 0, LQK, lane);
            v8f s = {};
            s = __builtin_amdgcn_wmma_f32_16x16x32_bf16(
                    false, a, false, kb, (short)0, s, false, false);
            int m0 = wv * 16 + ((lane >> 4) * 8);
            int n  = nt * 16 + (lane & 15);
            #pragma unroll
            for (int i = 0; i < 8; ++i) {
                int m = m0 + i;
                Ss[m * LSV + n] = f2bf(s[i] * dh[m * 80 + n]);
            }
        }
        for (int z = lane; z < 16 * 16; z += 32)               // zero score pad cols 80..95
            Ss[(wv * 16 + (z >> 4)) * LSV + 80 + (z & 15)] = bz;
    }
    __syncthreads();

    // O = S @ V : 16 rows x 32 cols, K = 96 (3 WMMA steps) per output-column tile
    v8f o[2] = {};
    #pragma unroll
    for (int ks = 0; ks < 3; ++ks) {
        v16bf a = lds_frag_a(Ss, wv * 16, ks * 32, LSV, lane);
        #pragma unroll
        for (int ni = 0; ni < 2; ++ni) {
            v16bf vb = lds_frag_b(Vt, ni * 16, ks * 32, LSV, lane);
            o[ni] = __builtin_amdgcn_wmma_f32_16x16x32_bf16(
                        false, a, false, vb, (short)0, o[ni], false, false);
        }
    }
    int m0 = wv * 16 + ((lane >> 4) * 8);
    #pragma unroll
    for (int ni = 0; ni < 2; ++ni) {
        int c = head * 32 + ni * 16 + (lane & 15);
        #pragma unroll
        for (int i = 0; i < 8; ++i) {
            int m = m0 + i;
            int row = (mode == 0) ? ((b * 80 + m) * 80 + fix)
                                  : ((b * 80 + fix) * 80 + m);
            out[(size_t)row * 256 + c] = o[ni][i];
        }
    }
}

// ---------------------------------------------------------------------------
// LayerNorm over C=256 -> bf16. One 256-thread block per row.
// ---------------------------------------------------------------------------
__global__ __launch_bounds__(256) void layernorm_bf16(
    const float* __restrict__ x, const float* __restrict__ g,
    const float* __restrict__ beta, __bf16* __restrict__ out)
{
    __shared__ float red[256];
    const int row = blockIdx.x;
    const int c   = threadIdx.x;
    float v = x[(size_t)row * 256 + c];
    red[c] = v; __syncthreads();
    for (int s = 128; s > 0; s >>= 1) { if (c < s) red[c] += red[c + s]; __syncthreads(); }
    float mu = red[0] * (1.0f / 256.0f);
    __syncthreads();
    float d = v - mu;
    red[c] = d * d; __syncthreads();
    for (int s = 128; s > 0; s >>= 1) { if (c < s) red[c] += red[c + s]; __syncthreads(); }
    float var = red[0] * (1.0f / 256.0f);
    float y = d * rsqrtf(var + 1e-5f) * g[c] + beta[c];
    out[(size_t)row * 256 + c] = f2bf(y);
}

// f32 W[K,N]  ->  bf16 Wt[N,K]  (one-time, per launch)
__global__ void transpose_to_bf16_kernel(const float* __restrict__ in,
                                         __bf16* __restrict__ out, int K, int N) {
    int i = blockIdx.x * 256 + threadIdx.x;
    if (i < K * N) {
        int k = i / N, n = i % N;
        out[(size_t)n * K + k] = f2bf(in[i]);
    }
}

__global__ void combine_attn_kernel(const float* __restrict__ a,
                                    const float* __restrict__ b,
                                    __bf16* __restrict__ out, int n) {
    int i = blockIdx.x * 256 + threadIdx.x;
    if (i < n) out[i] = f2bf(0.5f * (a[i] + b[i]));
}

// ---------------------------------------------------------------------------
extern "C" void kernel_launch(void* const* d_in, const int* in_sizes, int n_in,
                              void* d_out, int out_size, void* d_ws, size_t ws_size,
                              hipStream_t stream)
{
    const float* x       = (const float*)d_in[0];
    const float* decay_h = (const float*)d_in[1];
    const float* decay_w = (const float*)d_in[2];
    const float* qkv_w   = (const float*)d_in[3];
    const float* qkv_b   = (const float*)d_in[4];
    const float* proj_w  = (const float*)d_in[5];
    const float* proj_b  = (const float*)d_in[6];
    const float* ln1_g   = (const float*)d_in[7];
    const float* ln1_b   = (const float*)d_in[8];
    const float* ln2_g   = (const float*)d_in[9];
    const float* ln2_b   = (const float*)d_in[10];
    const float* ffn_w1  = (const float*)d_in[11];
    const float* ffn_b1  = (const float*)d_in[12];
    const float* ffn_w2  = (const float*)d_in[13];
    const float* ffn_b2  = (const float*)d_in[14];
    const float* gamma   = (const float*)d_in[15];
    float* out = (float*)d_out;

    const int M = 8 * 80 * 80;   // 51200 tokens
    const int C = 256;

    // ----- carve workspace (256B aligned chunks) -----
    char* p = (char*)d_ws;
    auto carve = [&](size_t bytes) -> char* {
        char* r = p; p += (bytes + 255) & ~(size_t)255; return r;
    };
    __bf16* wq   = (__bf16*)carve((size_t)256 * 768 * 2);   // qkv_w^T  bf16 [768][256]
    __bf16* wp   = (__bf16*)carve((size_t)256 * 256 * 2);   // proj_w^T bf16 [256][256]
    __bf16* w1   = (__bf16*)carve((size_t)256 * 512 * 2);   // ffn_w1^T bf16 [512][256]
    __bf16* w2   = (__bf16*)carve((size_t)512 * 256 * 2);   // ffn_w2^T bf16 [256][512]
    __bf16* h1   = (__bf16*)carve((size_t)M * C * 2);       // LN1 out
    __bf16* qkv  = (__bf16*)carve((size_t)M * 768 * 2);     // QKV out
    float*  oh   = (float*) carve((size_t)M * C * 4);       // row-axis retention out
    float*  ow   = (float*) carve((size_t)M * C * 4);       // col-axis retention out
    __bf16* attn = (__bf16*)carve((size_t)M * C * 2);       // combined attn bf16
    float*  x1   = (float*) carve((size_t)M * C * 4);       // x + proj*gamma
    __bf16* h2   = (__bf16*)carve((size_t)M * C * 2);       // LN2 out
    __bf16* mid  = (__bf16*)carve((size_t)M * 512 * 2);     // gelu(FFN1) out
    (void)ws_size; (void)in_sizes; (void)n_in; (void)out_size;

    // ----- weights f32 -> bf16, transposed to [N][K] -----
    transpose_to_bf16_kernel<<<(256 * 768 + 255) / 256, 256, 0, stream>>>(qkv_w, wq, 256, 768);
    transpose_to_bf16_kernel<<<(256 * 256 + 255) / 256, 256, 0, stream>>>(proj_w, wp, 256, 256);
    transpose_to_bf16_kernel<<<(256 * 512 + 255) / 256, 256, 0, stream>>>(ffn_w1, w1, 256, 512);
    transpose_to_bf16_kernel<<<(512 * 256 + 255) / 256, 256, 0, stream>>>(ffn_w2, w2, 512, 256);

    // ----- LN1 -----
    layernorm_bf16<<<M, 256, 0, stream>>>(x, ln1_g, ln1_b, h1);

    // ----- QKV GEMM: [51200,256] x [256,768] -----
    gemm_bf16_wmma<0><<<dim3(M / 64, 768 / 64), 128, 0, stream>>>(
        h1, wq, qkv_b, nullptr, nullptr, nullptr, qkv, M, 768, 256);

    // ----- retention: row pass (decay_h) and column pass (decay_w) -----
    retention_axis<<<8 * 8 * 80, 160, 0, stream>>>(qkv, decay_h, oh, 0);
    retention_axis<<<8 * 8 * 80, 160, 0, stream>>>(qkv, decay_w, ow, 1);
    combine_attn_kernel<<<(M * C + 255) / 256, 256, 0, stream>>>(oh, ow, attn, M * C);

    // ----- proj GEMM + residual*gamma: x1 = x + (attn @ proj_w + b) * gamma -----
    gemm_bf16_wmma<1><<<dim3(M / 64, 256 / 64), 128, 0, stream>>>(
        attn, wp, proj_b, x, gamma, x1, nullptr, M, 256, 256);

    // ----- LN2 -----
    layernorm_bf16<<<M, 256, 0, stream>>>(x1, ln2_g, ln2_b, h2);

    // ----- FFN1 GEMM + gelu -----
    gemm_bf16_wmma<2><<<dim3(M / 64, 512 / 64), 128, 0, stream>>>(
        h2, w1, ffn_b1, nullptr, nullptr, nullptr, mid, M, 512, 256);

    // ----- FFN2 GEMM + residual -> out -----
    gemm_bf16_wmma<3><<<dim3(M / 64, 256 / 64), 128, 0, stream>>>(
        mid, w2, ffn_b2, x1, nullptr, out, nullptr, M, 256, 512);
}
